// GatedGCNConv_60447369723960
// MI455X (gfx1250) — matrix operand order repaired and checked by reading, not compiled, not run on hardware
//
#include <hip/hip_runtime.h>

typedef __attribute__((ext_vector_type(16))) __bf16 v16bf;
typedef __attribute__((ext_vector_type(8)))  float  v8f;

#define DD 256
#define GEMM_TM 128
#define GEMM_THREADS 512
#define W_PITCH 264   // bf16 elements per LDS row: 256 + 8 pad (16B) -> kills bank conflicts

// LDS byte offset of a generic pointer known to alias LDS (addrspacecast to AS3).
__device__ __forceinline__ unsigned lds_off_u32(const void* p) {
  return (unsigned)(unsigned long long)(__attribute__((address_space(3))) const void*)p;
}

// ---------------------------------------------------------------------------
// Convert four fp32 quads into a 16-bit WMMA A-fragment.
// Lane layout (16-bit A 16x32): lanes 0-15 hold K=[0..7] in v[0..7], K=[16..23]
// in v[8..15]; lanes 16-31 hold K=[8..15] and K=[24..31]. Callers pass quads
// at (row + kb*32 + lh*8) + {0,4,16,20}.
// ---------------------------------------------------------------------------
__device__ __forceinline__ v16bf cvt_a_frag(float4 f0, float4 f1, float4 f2, float4 f3) {
  v16bf r;
  r[0]  = (__bf16)f0.x; r[1]  = (__bf16)f0.y; r[2]  = (__bf16)f0.z; r[3]  = (__bf16)f0.w;
  r[4]  = (__bf16)f1.x; r[5]  = (__bf16)f1.y; r[6]  = (__bf16)f1.z; r[7]  = (__bf16)f1.w;
  r[8]  = (__bf16)f2.x; r[9]  = (__bf16)f2.y; r[10] = (__bf16)f2.z; r[11] = (__bf16)f2.w;
  r[12] = (__bf16)f3.x; r[13] = (__bf16)f3.y; r[14] = (__bf16)f3.z; r[15] = (__bf16)f3.w;
  return r;
}

// ---------------------------------------------------------------------------
// Y[M x 256] = X[M x 256] @ W[256 x 256] + bias, fp32 in/out, bf16 WMMA math.
// Wt is the transposed bf16 weight: Wt[n][k] = W[k][n], DMA-staged fully in
// LDS (132 KB of the 320 KB WGP LDS) via global_load_async_to_lds_b128 and
// reused by 128 output rows per block. 512 threads = 16 waves: wave (wr, wn)
// computes rows [wr*16,+16) x cols [wn*128,+128) as 8 accumulator tiles over
// 8 K-steps of 32. Design point: 1 block/WGP (4 waves/SIMD) with deep
// per-wave pipelining — all 8 B-fragments of a K-step are fetched before the
// WMMA chain, and the next A slab is loaded raw (separate regs) during it.
// ---------------------------------------------------------------------------
__global__ void __launch_bounds__(GEMM_THREADS, 1)
k_gemm256_bf16(const float* __restrict__ X, const __bf16* __restrict__ Wt,
               const float* __restrict__ bias, float* __restrict__ Y, int M)
{
  extern __shared__ unsigned char smem[];
  __bf16* lw = (__bf16*)smem;

  const int tid = threadIdx.x;
  // Async DMA stage Wt -> padded LDS: 256 rows x 32 chunks of 16B, no VGPR
  // round trip (CDNA5 async path, tracked by ASYNCcnt).
  for (int c = tid; c < 256 * 32; c += GEMM_THREADS) {
    const int row = c >> 5, col8 = c & 31;
    const unsigned           ldst = lds_off_u32(lw + row * W_PITCH + col8 * 8);
    const unsigned long long ga   = (unsigned long long)(Wt + row * 256 + col8 * 8);
    asm volatile("global_load_async_to_lds_b128 %0, %1, off"
                 :: "v"(ldst), "v"(ga) : "memory");
  }
  asm volatile("s_wait_asynccnt 0x0" ::: "memory");
  __syncthreads();

  const int wave = tid >> 5;
  const int lane = tid & 31;
  const int lr   = lane & 15;   // N / M sub-index within half-wave
  const int lh   = lane >> 4;   // half-wave select
  const int wr   = wave & 7;    // row-tile group (0..7)
  const int wn   = wave >> 3;   // N half (0..1)
  const int mbase = blockIdx.x * GEMM_TM + wr * 16;

  long arow = mbase + lr;
  if (arow >= M) arow = M - 1;              // branchless clamp; stores masked below
  const float* xr = X + (size_t)arow * DD + lh * 8;

  v8f acc[8] = {};
  union BU { uint4 q[2]; v16bf v; };

  // Prologue: A(0).
  v16bf afrag;
  {
    const float4 f0 = *(const float4*)(xr + 0);
    const float4 f1 = *(const float4*)(xr + 4);
    const float4 f2 = *(const float4*)(xr + 16);
    const float4 f3 = *(const float4*)(xr + 20);
    afrag = cvt_a_frag(f0, f1, f2, f3);
  }

  for (int kb = 0; kb < 8; ++kb) {
    // Fetch ALL 8 B-fragments of this K-step up front (16 ds_load_b128 in
    // flight -> descending DScnt waits instead of lockstep wait-0).
    // B lane layout (bf16 32x16): lane = column N (lr), K = lh*8 + [0..7] in
    // v[0..7] and K = 16 + lh*8 + [0..7] in v[8..15] within the 32-slab.
    BU bb[8];
    const __bf16* lbase = lw + (size_t)(kb * 32 + lh * 8) + (size_t)lr * W_PITCH;
#pragma unroll
    for (int t = 0; t < 8; ++t) {
      const __bf16* bp = lbase + (size_t)((wn * 8 + t) * 16) * W_PITCH;
      bb[t].q[0] = *(const uint4*)(bp);        // ds_load_b128
      bb[t].q[1] = *(const uint4*)(bp + 16);   // ds_load_b128
    }
    // Next A slab: raw fp32 into separate registers; convert after the chain.
    float4 n0, n1, n2, n3;
    if (kb < 7) {
      const float* ap = xr + (kb + 1) * 32;
      n0 = *(const float4*)(ap + 0);
      n1 = *(const float4*)(ap + 4);
      n2 = *(const float4*)(ap + 16);
      n3 = *(const float4*)(ap + 20);
    }
    if (kb < 6) __builtin_prefetch(xr + (kb + 2) * 32, 0, 1);  // global_prefetch_b8
#pragma unroll
    for (int t = 0; t < 8; ++t) {
      acc[t] = __builtin_amdgcn_wmma_f32_16x16x32_bf16(
          false, afrag, false, bb[t].v, (short)0, acc[t], false, false);
    }
    if (kb < 7) afrag = cvt_a_frag(n0, n1, n2, n3);
  }

  // C/D layout: lane holds column N = lr, rows M = v + 8*lh for VGPR v.
#pragma unroll
  for (int t = 0; t < 8; ++t) {
    const int n = (wn * 8 + t) * 16 + lr;
    const float bv = bias[n];
#pragma unroll
    for (int v = 0; v < 8; ++v) {
      const int rs = mbase + v + 8 * lh;
      if (rs < M) Y[(size_t)rs * DD + n] = acc[t][v] + bv;
    }
  }
}

// ---------------------------------------------------------------------------
// Transpose+convert the 9 weights: Wt[w][n][k] = (bf16) W[w][k][n].
// ---------------------------------------------------------------------------
__global__ void k_wt(const float* __restrict__ W, __bf16* __restrict__ Wt) {
  const int idx = blockIdx.x * blockDim.x + threadIdx.x;
  if (idx >= 9 * 65536) return;
  const int w = idx >> 16;
  const int r = idx & 65535;
  const int n = r >> 8;
  const int k = r & 255;
  Wt[idx] = (__bf16)W[(w << 16) + (k << 8) + n];
}

__global__ void k_zero(float* __restrict__ p, size_t n) {
  size_t i = (size_t)blockIdx.x * blockDim.x + threadIdx.x;
  const size_t st = (size_t)gridDim.x * blockDim.x;
  for (; i < n; i += st) p[i] = 0.0f;
}

// e2pre = Ah[src] + Ah[dst] + Be + Cu[gid[src]]; accumulate column sum/sumsq.
__global__ void k_edge_pre(const float* __restrict__ Ah, const float* __restrict__ Be,
                           const float* __restrict__ Cu, const int* __restrict__ src,
                           const int* __restrict__ dst, const int* __restrict__ gid,
                           float* __restrict__ E2, float* __restrict__ stats, int NE) {
  const int c = threadIdx.x;                 // one column per thread
  const int r0 = blockIdx.x * 256;
  float s = 0.f, ss = 0.f;
  for (int i = 0; i < 256; ++i) {
    const int r = r0 + i;
    if (r >= NE) break;
    const int sa = src[r], da = dst[r];
    const int g = gid[sa];
    const float v = Ah[(size_t)sa * DD + c] + Ah[(size_t)da * DD + c] +
                    Be[(size_t)r * DD + c] + Cu[(size_t)g * DD + c];
    E2[(size_t)r * DD + c] = v;
    s += v; ss += v * v;
  }
  atomicAdd(&stats[c], s);
  atomicAdd(&stats[256 + c], ss);
}

// scale/shift from sums: scale = g*rsqrt(var+eps), shift = b - mean*scale.
__global__ void k_bn_params(const float* __restrict__ stats, const float* __restrict__ gamma,
                            const float* __restrict__ beta, float* __restrict__ prm,
                            float invN) {
  const int c = threadIdx.x;
  const float m = stats[c] * invN;
  const float var = stats[256 + c] * invN - m * m;
  const float sc = gamma[c] * rsqrtf(var + 1e-5f);
  prm[c] = sc;
  prm[256 + c] = beta[c] - m * sc;
}

__global__ void k_bn_relu(float* __restrict__ X, const float* __restrict__ prm, size_t n) {
  size_t i = (size_t)blockIdx.x * blockDim.x + threadIdx.x;
  const size_t st = (size_t)gridDim.x * blockDim.x;
  for (; i < n; i += st) {
    const int c = (int)(i & 255);
    const float v = X[i] * prm[c] + prm[256 + c];
    X[i] = v > 0.f ? v : 0.f;
  }
}

// Gated aggregation: num[dst] += sigmoid(e2)*Eh[src]; den[dst] += sigmoid(e2).
__global__ void k_agg(const float* __restrict__ E2, const float* __restrict__ Eh,
                      const int* __restrict__ src, const int* __restrict__ dst,
                      float* __restrict__ num, float* __restrict__ den, int NE) {
  const int c = threadIdx.x;
  const int r0 = blockIdx.x * 64;
  for (int i = 0; i < 64; ++i) {
    const int r = r0 + i;
    if (r >= NE) break;
    const int sa = src[r], da = dst[r];
    const float ev = E2[(size_t)r * DD + c];
    const float sg = 1.f / (1.f + __expf(-ev));
    atomicAdd(&num[(size_t)da * DD + c], sg * Eh[(size_t)sa * DD + c]);
    atomicAdd(&den[(size_t)da * DD + c], sg);
  }
}

// h2pre = Dh + num/(den+1e-6) + Fu[gid]; accumulate column stats.
__global__ void k_h2_pre(const float* __restrict__ Dh, const float* __restrict__ num,
                         const float* __restrict__ den, const float* __restrict__ Fu,
                         const int* __restrict__ gid, float* __restrict__ H2,
                         float* __restrict__ stats, int NA) {
  const int c = threadIdx.x;
  const int r0 = blockIdx.x * 256;
  float s = 0.f, ss = 0.f;
  for (int i = 0; i < 256; ++i) {
    const int r = r0 + i;
    if (r >= NA) break;
    const int g = gid[r];
    const size_t o = (size_t)r * DD + c;
    const float v = Dh[o] + num[o] / (den[o] + 1e-6f) + Fu[(size_t)g * DD + c];
    H2[o] = v;
    s += v; ss += v * v;
  }
  atomicAdd(&stats[c], s);
  atomicAdd(&stats[256 + c], ss);
}

// He_g[g] += He[edge] with g = gid[dst]; degg[g] += 1 per edge.
__global__ void k_heg(const float* __restrict__ He, const int* __restrict__ dst,
                      const int* __restrict__ gid, float* __restrict__ Heg,
                      float* __restrict__ degg, int NE) {
  const int c = threadIdx.x;
  const int r0 = blockIdx.x * 64;
  for (int i = 0; i < 64; ++i) {
    const int r = r0 + i;
    if (r >= NE) break;
    const int g = gid[dst[r]];
    atomicAdd(&Heg[(size_t)g * DD + c], He[(size_t)r * DD + c]);
    if (c == 0) atomicAdd(&degg[g], 1.f);
  }
}

// Ghs[g] += Gh[atom]; cnt[g] += 1 per atom.
__global__ void k_ghs(const float* __restrict__ Gh, const int* __restrict__ gid,
                      float* __restrict__ Ghs, float* __restrict__ cnt, int NA) {
  const int c = threadIdx.x;
  const int r0 = blockIdx.x * 64;
  for (int i = 0; i < 64; ++i) {
    const int r = r0 + i;
    if (r >= NA) break;
    const int g = gid[r];
    atomicAdd(&Ghs[(size_t)g * DD + c], Gh[(size_t)r * DD + c]);
    if (c == 0) atomicAdd(&cnt[g], 1.f);
  }
}

// u2pre = Ghs/cnt + Heg/degg + Iu; column stats.
__global__ void k_u2_pre(const float* __restrict__ Ghs, const float* __restrict__ cnt,
                         const float* __restrict__ Heg, const float* __restrict__ degg,
                         const float* __restrict__ Iu, float* __restrict__ U2,
                         float* __restrict__ stats, int NG) {
  const int c = threadIdx.x;
  const int r0 = blockIdx.x * 64;
  float s = 0.f, ss = 0.f;
  for (int i = 0; i < 64; ++i) {
    const int g = r0 + i;
    if (g >= NG) break;
    const size_t o = (size_t)g * DD + c;
    const float v = Ghs[o] / cnt[g] + Heg[o] / degg[g] + Iu[o];
    U2[o] = v;
    s += v; ss += v * v;
  }
  atomicAdd(&stats[c], s);
  atomicAdd(&stats[256 + c], ss);
}

// ---------------------------------------------------------------------------
extern "C" void kernel_launch(void* const* d_in, const int* in_sizes, int n_in,
                              void* d_out, int out_size, void* d_ws, size_t ws_size,
                              hipStream_t stream) {
  const float* h  = (const float*)d_in[0];
  const float* e  = (const float*)d_in[1];
  const float* u  = (const float*)d_in[2];
  const float* W  = (const float*)d_in[3];
  const float* b  = (const float*)d_in[4];
  const float* gm = (const float*)d_in[5];
  const float* bt = (const float*)d_in[6];
  const int* src  = (const int*)d_in[7];
  const int* dst  = (const int*)d_in[8];
  const int* gid  = (const int*)d_in[9];
  const int NA = in_sizes[0] / DD;
  const int NE = in_sizes[1] / DD;
  const int NG = in_sizes[2] / DD;

  // Workspace layout (regions re-used across stages):
  //   regE (NE*D): Be | {Dh, num, den} (3*NA <= NE) | He
  //   regH (NA*D): Ah | Eh | Gh
  //   regU (NG*D): Cu | Fu | Iu
  char* ws = (char*)d_ws;
  __bf16* Wt = (__bf16*)ws;
  size_t off = ((size_t)9 * 65536 * 2 + 255) & ~(size_t)255;
  float* regE  = (float*)(ws + off); off += (size_t)NE * DD * 4;
  float* regH  = (float*)(ws + off); off += (size_t)NA * DD * 4;
  float* regU  = (float*)(ws + off); off += (size_t)NG * DD * 4;
  float* stats = (float*)(ws + off); off += 3 * 512 * 4;   // [h|e|u] x (sum,sumsq)
  float* prm   = (float*)(ws + off); off += 3 * 512 * 4;   // [h|e|u] x (scale,shift)
  float* Heg   = (float*)(ws + off); off += (size_t)NG * DD * 4;
  float* Ghs   = (float*)(ws + off); off += (size_t)NG * DD * 4;
  float* degg  = (float*)(ws + off); off += (size_t)NG * 4;
  float* cnt   = (float*)(ws + off); off += (size_t)NG * 4;

  float* h2 = (float*)d_out;
  float* e2 = h2 + (size_t)NA * DD;
  float* u2 = e2 + (size_t)NE * DD;

  const size_t LDSB = (size_t)256 * W_PITCH * 2;   // 135168 B (<320 KB WGP LDS)
  const int gNA = (NA + GEMM_TM - 1) / GEMM_TM;
  const int gNE = (NE + GEMM_TM - 1) / GEMM_TM;
  const int gNG = (NG + GEMM_TM - 1) / GEMM_TM;

  k_wt<<<(9 * 65536 + 255) / 256, 256, 0, stream>>>(W, Wt);
  k_zero<<<64, 256, 0, stream>>>(stats, 3 * 512);

  // ---- stage 1: edge update ----
  k_gemm256_bf16<<<gNA, GEMM_THREADS, LDSB, stream>>>(h, Wt + 0 * 65536, b + 0 * DD, regH, NA); // Ah
  k_gemm256_bf16<<<gNE, GEMM_THREADS, LDSB, stream>>>(e, Wt + 1 * 65536, b + 1 * DD, regE, NE); // Be
  k_gemm256_bf16<<<gNG, GEMM_THREADS, LDSB, stream>>>(u, Wt + 2 * 65536, b + 2 * DD, regU, NG); // Cu
  k_edge_pre<<<(NE + 255) / 256, 256, 0, stream>>>(regH, regE, regU, src, dst, gid, e2, stats + 512, NE);
  k_bn_params<<<1, 256, 0, stream>>>(stats + 512, gm + DD, bt + DD, prm + 512, 1.0f / NE);
  k_bn_relu<<<4096, 256, 0, stream>>>(e2, prm + 512, (size_t)NE * DD);

  // ---- stage 2: atom update ----
  float* Dh  = regE;
  float* num = regE + (size_t)NA * DD;
  float* den = regE + (size_t)2 * NA * DD;
  k_gemm256_bf16<<<gNA, GEMM_THREADS, LDSB, stream>>>(h, Wt + 3 * 65536, b + 3 * DD, Dh,   NA); // Dh
  k_gemm256_bf16<<<gNA, GEMM_THREADS, LDSB, stream>>>(h, Wt + 4 * 65536, b + 4 * DD, regH, NA); // Eh
  k_gemm256_bf16<<<gNG, GEMM_THREADS, LDSB, stream>>>(u, Wt + 5 * 65536, b + 5 * DD, regU, NG); // Fu
  k_zero<<<4096, 256, 0, stream>>>(num, (size_t)2 * NA * DD);
  k_agg<<<(NE + 63) / 64, 256, 0, stream>>>(e2, regH, src, dst, num, den, NE);
  k_h2_pre<<<(NA + 255) / 256, 256, 0, stream>>>(Dh, num, den, regU, gid, h2, stats, NA);
  k_bn_params<<<1, 256, 0, stream>>>(stats, gm, bt, prm, 1.0f / NA);
  k_bn_relu<<<4096, 256, 0, stream>>>(h2, prm, (size_t)NA * DD);

  // ---- stage 3: global update ----
  k_gemm256_bf16<<<gNA, GEMM_THREADS, LDSB, stream>>>(h2, Wt + 6 * 65536, b + 6 * DD, regH, NA); // Gh
  k_gemm256_bf16<<<gNE, GEMM_THREADS, LDSB, stream>>>(e2, Wt + 7 * 65536, b + 7 * DD, regE, NE); // He
  k_gemm256_bf16<<<gNG, GEMM_THREADS, LDSB, stream>>>(u,  Wt + 8 * 65536, b + 8 * DD, regU, NG); // Iu
  k_zero<<<64, 256, 0, stream>>>(Heg, (size_t)2 * NG * DD + 2 * NG);  // Heg,Ghs,degg,cnt
  k_heg<<<(NE + 63) / 64, 256, 0, stream>>>(regE, dst, gid, Heg, degg, NE);
  k_ghs<<<(NA + 63) / 64, 256, 0, stream>>>(regH, gid, Ghs, cnt, NA);
  k_u2_pre<<<(NG + 63) / 64, 256, 0, stream>>>(Ghs, cnt, Heg, degg, regU, u2, stats + 1024, NG);
  k_bn_params<<<1, 256, 0, stream>>>(stats + 1024, gm + 2 * DD, bt + 2 * DD, prm + 1024, 1.0f / NG);
  k_bn_relu<<<64, 256, 0, stream>>>(u2, prm + 1024, (size_t)NG * DD);
}
